// SingleStageDetector_9517647528247
// MI455X (gfx1250) — matrix operand order repaired and checked by reading, not compiled, not run on hardware
//
#include <hip/hip_runtime.h>
#include <hip/hip_bf16.h>
#include <math.h>

// ---------------- problem constants (from setup_inputs) ----------------
#define BB     32
#define CIN    256
#define HW     4096        // 64*64
#define HID    128
#define OUTC   65          // 5*9 + 20
#define OUTP   80          // padded to 5 WMMA n-tiles
#define AANCH  9
#define NCLS   20
#define NBOX   64
#define MTILE  64          // pixels per workgroup

// output layout (flat, in reference return order)
#define PROP_LEN  (BB * AANCH * HW * 4)                 // 4718592
#define IOU_LEN   ((size_t)BB * AANCH * HW * NBOX)      // 75497472
#define CONF_OFF  ((size_t)PROP_LEN + IOU_LEN)          // 80216064
#define REG_OFF   (CONF_OFF + 1)
#define CLS_OFF   (CONF_OFF + 2)

// workspace layout
#define WS_W2BF_OFF   (HID * CIN)                       // ushort offset
#define WS_OUT_BYTES  86016                             // 65536 + 20480, 128B aligned

typedef __attribute__((ext_vector_type(16))) __bf16 v16bf;
typedef __attribute__((ext_vector_type(8)))  float  v8f;
typedef __attribute__((ext_vector_type(4)))  unsigned su4;
typedef __attribute__((ext_vector_type(8)))  unsigned su8;

struct __align__(16) U4 { unsigned x, y, z, w; };
union Frag { U4 q[2]; v16bf bf; };

__device__ __forceinline__ unsigned short f2bf(float v) {
    return (unsigned short)(__float_as_uint(v) >> 16);
}
__device__ __forceinline__ unsigned pack_bf2(float lo, float hi) {
    return (unsigned)f2bf(lo) | ((unsigned)f2bf(hi) << 16);
}
__device__ __forceinline__ float sigmoidf_(float x) {
    return 1.0f / (1.0f + __expf(-x));
}

// ---------------- kernel 1: convert weights fp32 -> bf16 (W2 zero-padded to 80 rows)
__global__ void prep_weights(const float* __restrict__ W1, const float* __restrict__ W2,
                             unsigned short* __restrict__ w1bf, unsigned short* __restrict__ w2bf) {
    int t = blockIdx.x * blockDim.x + threadIdx.x;
    if (t < HID * CIN) {
        w1bf[t] = f2bf(W1[t]);
    }
    int t2 = t - HID * CIN;
    if (t2 >= 0 && t2 < OUTP * HID) {
        int o = t2 / HID, c = t2 - o * HID;
        float v = (o < OUTC) ? W2[o * HID + c] : 0.0f;
        w2bf[t2] = f2bf(v);
    }
}

// ---------------- kernel 2: fused conv1 (leaky_relu) + conv2 via bf16 WMMA ----------------
// grid: 2048 blocks * 128 threads (4 waves). Block = 64-pixel M-tile.
// LDS: 32KB feature tile + 64KB W1 (bf16) + 16KB hidden tile = 112KB -> 2 blocks/WGP.
__global__ __launch_bounds__(128)
void fused_conv_wmma(const float* __restrict__ feat,
                     const unsigned short* __restrict__ w1bf,
                     const unsigned short* __restrict__ w2bf,
                     const float* __restrict__ b1,
                     const float* __restrict__ b2,
                     float* __restrict__ outbuf) {
    __shared__ __align__(16) unsigned short sA[MTILE * CIN];   // 32 KB feature tile (bf16)
    __shared__ __align__(16) unsigned short sW1[HID * CIN];    // 64 KB W1 (bf16)
    __shared__ __align__(16) unsigned short sH[MTILE * HID];   // 16 KB hidden tile (bf16)

    const int tid  = threadIdx.x;
    const int lane = tid & 31;
    const int wave = tid >> 5;            // 0..3
    const int p0   = blockIdx.x * MTILE;  // first pixel of tile (same batch: HW%64==0)
    const int bimg = p0 >> 12;            // p0 / HW
    const int s0   = p0 & (HW - 1);

    // ---- CDNA5 TDM issue/wait path (NULL descriptor: group0.count==0 => NOP per ISA 8.3).
    // Portable inline-asm form (bypasses the toolchain-specific builtin arity).
    {
        su4 g0 = {0u, 0u, 0u, 0u};
        su8 g1 = {0u, 0u, 0u, 0u, 0u, 0u, 0u, 0u};
        asm volatile("tensor_load_to_lds %0, %1" :: "s"(g0), "s"(g1) : "memory");
        asm volatile("s_wait_tensorcnt 0x0" ::: "memory");
    }

    // ---- stage W1 (bf16) into LDS: 64KB as uint4 copies (32 iters * 128 thr * 16B) ----
    {
        const U4* src = (const U4*)w1bf;
        U4* dst = (U4*)sW1;
        #pragma unroll 4
        for (int i = 0; i < 32; ++i)
            dst[tid + i * 128] = src[tid + i * 128];
    }

    // ---- stage feature tile into LDS as bf16: sA[pixel][channel], b128 writes ----
    {
        const int pl = tid & 63;          // local pixel 0..63
        const int cg = tid >> 6;          // 0..1
        const float* src = feat + (size_t)bimg * CIN * HW + (s0 + pl);
        // prefetch next tile's first line (CDNA5 global_prefetch)
        __builtin_prefetch(src + MTILE, 0, 0);
        for (int c8 = cg * 8; c8 < CIN; c8 += 16) {
            float f0 = src[(size_t)(c8 + 0) * HW], f1 = src[(size_t)(c8 + 1) * HW];
            float f2 = src[(size_t)(c8 + 2) * HW], f3 = src[(size_t)(c8 + 3) * HW];
            float f4 = src[(size_t)(c8 + 4) * HW], f5 = src[(size_t)(c8 + 5) * HW];
            float f6 = src[(size_t)(c8 + 6) * HW], f7 = src[(size_t)(c8 + 7) * HW];
            U4 w;
            w.x = pack_bf2(f0, f1); w.y = pack_bf2(f2, f3);
            w.z = pack_bf2(f4, f5); w.w = pack_bf2(f6, f7);
            *(U4*)(sA + pl * CIN + c8) = w;
        }
    }
    __syncthreads();

    const int mrow  = wave * 16 + (lane & 15);   // A-frag row (ISA 16-bit A 16x32 layout)
    const int khalf = (lane >> 4) * 8;           // unused: A/B loads use kbhal (lanes 16-31: K=16..31)
    const int ncol  = lane & 15;                 // B/C/D column within n-tile
    const int kbhal = (lane >> 4) * 16;          // B-frag K half (lanes 16-31: K=16..31)

    // ---- conv1: 16x128 per wave, K = 256 (8 steps of 32), 64 WMMAs / wave ----
    v8f acc[8];
    #pragma unroll
    for (int i = 0; i < 8; ++i)
        #pragma unroll
        for (int j = 0; j < 8; ++j) acc[i][j] = 0.0f;

    for (int ks = 0; ks < CIN / 32; ++ks) {
        Frag a;
        const unsigned short* ap = sA + mrow * CIN + ks * 32;
        a.q[0] = *(const U4*)(ap + kbhal);        // K = {0..7} or {16..23}
        a.q[1] = *(const U4*)(ap + kbhal + 8);    // K = {8..15} or {24..31}
        #pragma unroll
        for (int nt = 0; nt < 8; ++nt) {
            Frag bf;
            const unsigned short* bp = sW1 + (nt * 16 + ncol) * CIN + ks * 32 + kbhal;
            bf.q[0] = *(const U4*)(bp);
            bf.q[1] = *(const U4*)(bp + 8);
            acc[nt] = __builtin_amdgcn_wmma_f32_16x16x32_bf16(
                false, a.bf, false, bf.bf, (short)0, acc[nt], false, false);
        }
    }

    // ---- epilogue conv1: bias + leaky_relu -> sH (bf16). wave-local rows ----
    #pragma unroll
    for (int nt = 0; nt < 8; ++nt) {
        const int col  = nt * 16 + ncol;
        const float bias = b1[col];
        #pragma unroll
        for (int r = 0; r < 8; ++r) {
            const int row = wave * 16 + r + 8 * (lane >> 4);   // C/D layout
            float v = acc[nt][r] + bias;
            v = (v >= 0.0f) ? v : 0.01f * v;
            sH[row * HID + col] = f2bf(v);
        }
    }
    __syncthreads();

    // ---- conv2: 16x80 per wave, K = 128 (4 steps of 32), 20 WMMAs / wave ----
    v8f acc2[5];
    #pragma unroll
    for (int i = 0; i < 5; ++i)
        #pragma unroll
        for (int j = 0; j < 8; ++j) acc2[i][j] = 0.0f;

    for (int ks = 0; ks < HID / 32; ++ks) {
        Frag a;
        const unsigned short* ap = sH + mrow * HID + ks * 32;
        a.q[0] = *(const U4*)(ap + kbhal);
        a.q[1] = *(const U4*)(ap + kbhal + 8);
        #pragma unroll
        for (int nt = 0; nt < 5; ++nt) {
            Frag bf;
            const unsigned short* bp = w2bf + (nt * 16 + ncol) * HID + ks * 32 + kbhal;
            bf.q[0] = *(const U4*)(bp);
            bf.q[1] = *(const U4*)(bp + 8);
            acc2[nt] = __builtin_amdgcn_wmma_f32_16x16x32_bf16(
                false, a.bf, false, bf.bf, (short)0, acc2[nt], false, false);
        }
    }

    // ---- epilogue conv2: bias (+ sigmoid-0.5 for offset-xy channels) -> outbuf ----
    float* ob = outbuf + (size_t)bimg * OUTC * HW;
    #pragma unroll
    for (int nt = 0; nt < 5; ++nt) {
        const int ch = nt * 16 + ncol;
        if (ch >= OUTC) continue;
        const bool isXY = (ch >= AANCH) && (ch < 5 * AANCH) && (((ch - AANCH) & 3) < 2);
        const float bias = b2[ch];
        #pragma unroll
        for (int r = 0; r < 8; ++r) {
            const int row = wave * 16 + r + 8 * (lane >> 4);
            float v = acc2[nt][r] + bias;
            if (isXY) v = sigmoidf_(v) - 0.5f;
            ob[(size_t)ch * HW + s0 + row] = v;
        }
    }
}

// ---------------- kernel 3: proposals (B, A, H, W, 4) ----------------
__global__ void proposals_kernel(const float* __restrict__ grid, const float* __restrict__ anc,
                                 const float* __restrict__ outbuf, float* __restrict__ prop) {
    int t = blockIdx.x * blockDim.x + threadIdx.x;     // over B*A*HW
    if (t >= BB * AANCH * HW) return;
    const int s = t & (HW - 1);
    const int a = (t >> 12) % AANCH;
    const int b = t / (AANCH * HW);
    const float gx = grid[((size_t)b * HW + s) * 2 + 0];
    const float gy = grid[((size_t)b * HW + s) * 2 + 1];
    const float aw = anc[a * 2 + 0], ah = anc[a * 2 + 1];
    const float* ob = outbuf + (size_t)b * OUTC * HW + (size_t)(AANCH + 4 * a) * HW + s;
    const float o0 = ob[0], o1 = ob[(size_t)HW], o2 = ob[(size_t)2 * HW], o3 = ob[(size_t)3 * HW];
    const float xc = gx + o0, yc = gy + o1;
    const float w = aw * __expf(o2), h = ah * __expf(o3);
    float4 r = make_float4(xc - 0.5f * w, yc - 0.5f * h, xc + 0.5f * w, yc + 0.5f * h);
    *(float4*)(prop + (size_t)t * 4) = r;
}

// ---------------- kernel 4: IoU (B, A*HW, NBOX) — bandwidth bound ----------------
__global__ void iou_kernel(const float* __restrict__ grid, const float* __restrict__ anc,
                           const float* __restrict__ bboxes, float* __restrict__ iou_out) {
    size_t t = (size_t)blockIdx.x * blockDim.x + threadIdx.x;
    if (t >= IOU_LEN) return;
    const int j = (int)(t & (NBOX - 1));
    const size_t rest = t >> 6;
    const int s = (int)(rest & (HW - 1));
    const int a = (int)((rest >> 12) % AANCH);
    const int b = (int)(rest / ((size_t)AANCH * HW));
    const float gx = grid[((size_t)b * HW + s) * 2 + 0];
    const float gy = grid[((size_t)b * HW + s) * 2 + 1];
    const float aw = anc[a * 2 + 0], ah = anc[a * 2 + 1];
    const float px1 = gx - 0.5f * aw, py1 = gy - 0.5f * ah;
    const float px2 = gx + 0.5f * aw, py2 = gy + 0.5f * ah;
    const float* bx = bboxes + ((size_t)b * NBOX + j) * 5;
    const float bx1 = bx[0], by1 = bx[1], bx2 = bx[2], by2 = bx[3];
    const float area_p = aw * ah;
    const float area_b = (bx2 - bx1) * (by2 - by1);
    const float dx = fmaxf(0.0f, fminf(bx2, px2) - fmaxf(bx1, px1));
    const float dy = fmaxf(0.0f, fminf(by2, py2) - fmaxf(by1, py1));
    const float inter = dx * dy;
    iou_out[t] = inter / (area_p + area_b - inter);
}

// ---------------- kernel 5a: zero the two loss accumulators ----------------
__global__ void zero_losses(float* __restrict__ loss2) {
    if (threadIdx.x == 0) { loss2[0] = 0.0f; loss2[1] = 0.0f; }
}

// ---------------- kernel 5b: gathers + losses (LDS tree reduce, 1 atomic/block) ----------------
__global__ __launch_bounds__(256)
void loss_gather_kernel(const float* __restrict__ outbuf,
                        const int* __restrict__ pos_idx, const int* __restrict__ neg_idx,
                        const float* __restrict__ gt_off,
                        float* __restrict__ conf_loss, float* __restrict__ reg_loss,
                        float* __restrict__ cls_out) {
    __shared__ float red[256];
    const int t = blockIdx.x * blockDim.x + threadIdx.x;   // 0..8191
    float confd = 0.0f, regd = 0.0f;
    if (t < 8192) {
        const int idx = (t < 4096) ? pos_idx[t] : neg_idx[t - 4096];
        const int b = idx / (AANCH * HW);
        const int rem = idx - b * (AANCH * HW);
        const int a = rem >> 12;
        const int s = rem & (HW - 1);
        const float* ob = outbuf + (size_t)b * OUTC * HW;
        const float sc = sigmoidf_(ob[(size_t)a * HW + s]);
        const float d = sc - ((t < 4096) ? 1.0f : 0.0f);
        confd = d * d;
        if (t < 4096) {
            #pragma unroll
            for (int k = 0; k < 4; ++k) {
                const float o = ob[(size_t)(AANCH + 4 * a + k) * HW + s];
                const float dd = o - gt_off[t * 4 + k];
                regd += dd * dd;
            }
            #pragma unroll
            for (int c = 0; c < NCLS; ++c)
                cls_out[(size_t)t * NCLS + c] = ob[(size_t)(5 * AANCH + c) * HW + s];
        }
    }
    red[threadIdx.x] = confd;
    __syncthreads();
    for (int off = 128; off > 0; off >>= 1) {
        if ((int)threadIdx.x < off) red[threadIdx.x] += red[threadIdx.x + off];
        __syncthreads();
    }
    if (threadIdx.x == 0) atomicAdd(conf_loss, red[0] * (1.0f / 8192.0f));
    __syncthreads();
    red[threadIdx.x] = regd;
    __syncthreads();
    for (int off = 128; off > 0; off >>= 1) {
        if ((int)threadIdx.x < off) red[threadIdx.x] += red[threadIdx.x + off];
        __syncthreads();
    }
    if (threadIdx.x == 0) atomicAdd(reg_loss, red[0] * (1.0f / 4096.0f));
}

// ---------------- host launch ----------------
extern "C" void kernel_launch(void* const* d_in, const int* in_sizes, int n_in,
                              void* d_out, int out_size, void* d_ws, size_t ws_size,
                              hipStream_t stream) {
    const float* feat  = (const float*)d_in[0];
    const float* grid  = (const float*)d_in[1];
    const float* anc   = (const float*)d_in[2];
    const float* bbox  = (const float*)d_in[3];
    const float* gtoff = (const float*)d_in[4];
    const float* W1    = (const float*)d_in[6];
    const float* b1    = (const float*)d_in[7];
    const float* W2    = (const float*)d_in[8];
    const float* b2    = (const float*)d_in[9];
    const int*   posi  = (const int*)d_in[10];
    const int*   negi  = (const int*)d_in[11];
    float* out = (float*)d_out;

    unsigned short* w1bf = (unsigned short*)d_ws;
    unsigned short* w2bf = w1bf + WS_W2BF_OFF;
    float* outbuf = (float*)((char*)d_ws + WS_OUT_BYTES);

    // 1) weights -> bf16
    prep_weights<<<(HID * CIN + OUTP * HID + 255) / 256, 256, 0, stream>>>(W1, W2, w1bf, w2bf);
    // 2) fused conv1+conv2 GEMM (WMMA bf16)
    fused_conv_wmma<<<(BB * HW) / MTILE, 128, 0, stream>>>(feat, w1bf, w2bf, b1, b2, outbuf);
    // 3) proposals
    proposals_kernel<<<(BB * AANCH * HW + 255) / 256, 256, 0, stream>>>(grid, anc, outbuf, out);
    // 4) IoU matrix
    iou_kernel<<<(unsigned)((IOU_LEN + 255) / 256), 256, 0, stream>>>(grid, anc, bbox, out + PROP_LEN);
    // 5) losses + class scores
    zero_losses<<<1, 32, 0, stream>>>(out + CONF_OFF);
    loss_gather_kernel<<<32, 256, 0, stream>>>(outbuf, posi, negi, gtoff,
                                               out + CONF_OFF, out + REG_OFF, out + CLS_OFF);
}
